// RelAttnConv_47450798686731
// MI455X (gfx1250) — matrix operand order repaired and checked by reading, not compiled, not run on hardware
//
#include <hip/hip_runtime.h>
#include <hip/hip_bf16.h>
#include <math.h>

typedef __attribute__((ext_vector_type(16))) _Float16 v16h;
typedef __attribute__((ext_vector_type(8)))  _Float16 v8h;
typedef __attribute__((ext_vector_type(8)))  float    v8f;

#define NNODES 100000
#define NEDGES 600000
#define DIM    164
#define NHEAD  2
#define DK     82
#define EDIM   16
#define TDIM   8
#define STEP   30000
#define NCHUNK 20

// fast tanh: exact at +/-inf, single v_exp_f32 transcendental
__device__ __forceinline__ float fast_tanh(float x) {
    return 1.0f - 2.0f / (__expf(2.0f * x) + 1.0f);
}

// ---------------------------------------------------------------- zero out
__global__ void zero_kernel(float* p, int n) {
    int i = blockIdx.x * blockDim.x + threadIdx.x;
    int stride = gridDim.x * blockDim.x;
    for (; i < n; i += stride) p[i] = 0.0f;
}

// ------------------------------------------------- pad/convert weights f16
// W1h: [96 n][32 k], zero outside (n<82, k<24).  W2h: [336 n][96 k], zero
// outside (n<328, k<82).  Zero padding makes all WMMA tiles exact.
__global__ void prep_weights(const float* __restrict__ W1,
                             const float* __restrict__ W2,
                             _Float16* __restrict__ W1h,
                             _Float16* __restrict__ W2h) {
    int i = blockIdx.x * blockDim.x + threadIdx.x;
    if (i < 96 * 32) {
        int n = i >> 5, k = i & 31;
        W1h[i] = (n < 82 && k < 24) ? (_Float16)W1[n * 24 + k] : (_Float16)0.0f;
    }
    if (i < 336 * 96) {
        int n = i / 96, k = i - n * 96;
        W2h[i] = (n < 328 && k < 82) ? (_Float16)W2[n * 82 + k] : (_Float16)0.0f;
    }
}

// ---------------------------------------------------- attention raw scores
// one wave32 per edge; coalesced row loads, shfl butterfly reduce
__global__ void score_kernel(const float* __restrict__ q,
                             const float* __restrict__ k,
                             const int* __restrict__ ei,
                             const float* __restrict__ rel_bias,
                             float* __restrict__ scorebuf) {
    int warp = (blockIdx.x * blockDim.x + threadIdx.x) >> 5;
    int lane = threadIdx.x & 31;
    if (warp >= NEDGES) return;
    int s = ei[warp];            // src
    int d = ei[NEDGES + warp];   // dst
    const float* qr = q + (size_t)d * DIM;
    const float* kr = k + (size_t)s * DIM;
    float a0 = 0.0f, a1 = 0.0f;
    #pragma unroll
    for (int t = 0; t < 6; t++) {
        int idx = lane + 32 * t;
        if (idx < DIM) {
            float p = qr[idx] * kr[idx];
            if (idx < DK) a0 += p; else a1 += p;
        }
    }
    #pragma unroll
    for (int m = 16; m >= 1; m >>= 1) {
        a0 += __shfl_xor(a0, m, 32);
        a1 += __shfl_xor(a1, m, 32);
    }
    if (lane == 0) {
        const float inv = 0.110431526074847f; // 1/sqrt(82)
        scorebuf[warp * 2 + 0] = a0 * inv + rel_bias[0];
        scorebuf[warp * 2 + 1] = a1 * inv + rel_bias[1];
    }
}

// ------------------------------------------------- per-(chunk,head) reduce
__global__ void chunkmax_kernel(const float* __restrict__ scorebuf,
                                float* __restrict__ chmax) {
    __shared__ float red[256];
    int c = blockIdx.x / NHEAD, h = blockIdx.x % NHEAD;
    float mx = -3.0e38f;
    for (int i = threadIdx.x; i < STEP; i += blockDim.x) {
        float s = scorebuf[((size_t)c * STEP + i) * NHEAD + h];
        mx = fmaxf(mx, s);
    }
    red[threadIdx.x] = mx;
    __syncthreads();
    for (int w = 128; w >= 1; w >>= 1) {
        if (threadIdx.x < w) red[threadIdx.x] = fmaxf(red[threadIdx.x], red[threadIdx.x + w]);
        __syncthreads();
    }
    if (threadIdx.x == 0) chmax[blockIdx.x] = red[0];
}

__global__ void chunksum_kernel(float* __restrict__ scorebuf,
                                const float* __restrict__ chmax,
                                float* __restrict__ chsum) {
    __shared__ float red[256];
    int c = blockIdx.x / NHEAD, h = blockIdx.x % NHEAD;
    float mx = chmax[blockIdx.x];
    float acc = 0.0f;
    for (int i = threadIdx.x; i < STEP; i += blockDim.x) {
        size_t idx = ((size_t)c * STEP + i) * NHEAD + h;
        float ex = __expf(scorebuf[idx] - mx);
        scorebuf[idx] = ex;   // buffer now holds exp values
        acc += ex;
    }
    red[threadIdx.x] = acc;
    __syncthreads();
    for (int w = 128; w >= 1; w >>= 1) {
        if (threadIdx.x < w) red[threadIdx.x] += red[threadIdx.x + w];
        __syncthreads();
    }
    if (threadIdx.x == 0) chsum[blockIdx.x] = red[0];
}

// --------------------------------------------------- FiLM MLP (WMMA) + apply
// Each wave owns a 16-edge tile and a private LDS staging buffer (16x328 f16)
// used first for hmid (stride 96) and then for tanh(gb) (stride 328).
__global__ __launch_bounds__(128) void film_kernel(
    const float* __restrict__ vmat,
    const int* __restrict__ ei,
    const float* __restrict__ edge_attr,
    const float* __restrict__ edge_time,
    const float* __restrict__ Wt,
    const float* __restrict__ bt,
    const float* __restrict__ b1,
    const float* __restrict__ b2,
    const _Float16* __restrict__ W1h,
    const _Float16* __restrict__ W2h,
    const float* __restrict__ exbuf,
    const float* __restrict__ chsum,
    float* __restrict__ out) {
    __shared__ _Float16 stag[4][16 * 328];   // 41984 bytes

    int wv = threadIdx.x >> 5;
    int lane = threadIdx.x & 31;
    int e0 = (blockIdx.x * 4 + wv) * 16;
    int m = lane & 15;
    bool hi = lane >= 16;
    int e = e0 + m;

    // ---- A-matrix fragment: feat[16 edges x 32k] (k<24 valid) -------------
    // lanes 0-15 : halves 0-7 -> k=0..7,  halves 8-15 -> k=16..23
    // lanes 16-31: halves 0-7 -> k=8..15, halves 8-15 -> k=24..31 (zeros)
    v16h afeat;
    {
        const float* ea = edge_attr + (size_t)e * EDIM + (hi ? 8 : 0);
        #pragma unroll
        for (int i = 0; i < 8; i++) afeat[i] = (_Float16)ea[i];
        if (!hi) {
            float t = edge_time[e];
            #pragma unroll
            for (int i = 0; i < 8; i++) afeat[8 + i] = (_Float16)(t * Wt[i] + bt[i]);
        } else {
            #pragma unroll
            for (int i = 0; i < 8; i++) afeat[8 + i] = (_Float16)0.0f;
        }
    }

    // ---- GEMM1: hmid[16 x 96] = feat x W1^T, bias + ReLU, stage to LDS ----
    _Float16* hS = &stag[wv][0];   // stride 96
    #pragma unroll
    for (int nt = 0; nt < 6; nt++) {
        int n = nt * 16 + m;
        const _Float16* bp = W1h + (size_t)n * 32 + (hi ? 8 : 0);
        v16h b = __builtin_shufflevector(*(const v8h*)bp, *(const v8h*)(bp + 16), 0, 1, 2, 3, 4, 5, 6, 7,
                                         8, 9, 10, 11, 12, 13, 14, 15);
        float bias = (n < 82) ? b1[n] : 0.0f;
        v8f c;
        #pragma unroll
        for (int r = 0; r < 8; r++) c[r] = bias;
        c = __builtin_amdgcn_wmma_f32_16x16x32_f16(false, afeat, false, b,
                                                   (short)0, c, false, false);
        int mm = hi ? 8 : 0;
        #pragma unroll
        for (int r = 0; r < 8; r++) {
            float hval = c[r] > 0.0f ? c[r] : 0.0f;
            hS[(mm + r) * 96 + n] = (_Float16)hval;
        }
    }
    asm volatile("s_wait_dscnt 0x0" ::: "memory");

    // ---- re-load hmid as 3 A-fragments (16x32 each) -----------------------
    v16h ah[3];
    #pragma unroll
    for (int kc = 0; kc < 3; kc++) {
        int run0 = kc * 32 + (hi ? 8 : 0);
        v8h lo = *(const v8h*)(hS + m * 96 + run0);
        v8h hh = *(const v8h*)(hS + m * 96 + run0 + 16);
        ah[kc] = __builtin_shufflevector(lo, hh, 0, 1, 2, 3, 4, 5, 6, 7,
                                         8, 9, 10, 11, 12, 13, 14, 15);
    }

    // ---- GEMM2: gb[16 x 328] = hmid x W2^T, bias + tanh, stage to LDS -----
    _Float16* gS = &stag[wv][0];   // stride 328 (overwrites hmid; safe)
    #pragma unroll
    for (int nt = 0; nt < 21; nt++) {
        int n = nt * 16 + m;       // 0..335 (W2h padded to 336)
        float bias = (n < 328) ? b2[n] : 0.0f;
        v8f c;
        #pragma unroll
        for (int r = 0; r < 8; r++) c[r] = bias;
        #pragma unroll
        for (int kc = 0; kc < 3; kc++) {
            const _Float16* bp = W2h + (size_t)n * 96 + kc * 32 + (hi ? 8 : 0);
            v16h b = __builtin_shufflevector(*(const v8h*)bp, *(const v8h*)(bp + 16), 0, 1, 2, 3, 4, 5, 6, 7,
                                             8, 9, 10, 11, 12, 13, 14, 15);
            c = __builtin_amdgcn_wmma_f32_16x16x32_f16(false, ah[kc], false, b,
                                                       (short)0, c, false, false);
        }
        if (n < 328) {
            int mm = hi ? 8 : 0;
            #pragma unroll
            for (int r = 0; r < 8; r++)
                gS[(mm + r) * 328 + n] = (_Float16)fast_tanh(c[r]);
        }
    }
    asm volatile("s_wait_dscnt 0x0" ::: "memory");

    // ---- apply: out[dst] += att * (v[src]*(1+gamma)+beta) -----------------
    const int* src = ei;
    const int* dst = ei + NEDGES;
    int c = e0 / STEP;                 // tile never straddles a chunk
    float inv0 = 1.0f / chsum[c * 2 + 0];
    float inv1 = 1.0f / chsum[c * 2 + 1];
    for (int mi = 0; mi < 16; mi++) {
        int ee = e0 + mi;
        int ss = src[ee], dd = dst[ee];
        float a0 = exbuf[ee * 2 + 0] * inv0;
        float a1 = exbuf[ee * 2 + 1] * inv1;
        const float* vr = vmat + (size_t)ss * DIM;
        float* orow = out + (size_t)dd * DIM;
        #pragma unroll
        for (int t = 0; t < 6; t++) {
            int d = lane + 32 * t;
            if (d < DIM) {
                float g  = (float)gS[mi * 328 + d];
                float bb = (float)gS[mi * 328 + DIM + d];
                float vm = vr[d] * (1.0f + g) + bb;
                float att = (d < DK) ? a0 : a1;
                atomicAdd(&orow[d], att * vm);
            }
        }
    }
}

// ---------------------------------------------------------------- launcher
extern "C" void kernel_launch(void* const* d_in, const int* in_sizes, int n_in,
                              void* d_out, int out_size, void* d_ws, size_t ws_size,
                              hipStream_t stream) {
    const float* q         = (const float*)d_in[0];
    const float* k         = (const float*)d_in[1];
    const float* v         = (const float*)d_in[2];
    const int*   ei        = (const int*)d_in[3];
    const float* edge_attr = (const float*)d_in[4];
    const float* edge_time = (const float*)d_in[5];
    const float* Wt        = (const float*)d_in[6];
    const float* bt        = (const float*)d_in[7];
    const float* W1        = (const float*)d_in[8];
    const float* b1        = (const float*)d_in[9];
    const float* W2        = (const float*)d_in[10];
    const float* b2        = (const float*)d_in[11];
    const float* rel_bias  = (const float*)d_in[12];
    float* out = (float*)d_out;

    // workspace layout (all offsets 256B-aligned)
    char* w = (char*)d_ws;
    float* exbuf = (float*)w;                                  // E*2 f32 = 4.8MB
    size_t off = (size_t)NEDGES * 2 * sizeof(float);           // 4800000 (256-mult)
    float* chmax = (float*)(w + off); off += 256;
    float* chsum = (float*)(w + off); off += 256;
    _Float16* W1h = (_Float16*)(w + off); off += 96 * 32 * sizeof(_Float16);
    _Float16* W2h = (_Float16*)(w + off);

    // 1. zero output accumulator
    zero_kernel<<<2048, 256, 0, stream>>>(out, NNODES * DIM);
    // 2. pad/convert weights to f16 in workspace
    prep_weights<<<(336 * 96 + 255) / 256, 256, 0, stream>>>(W1, W2, W1h, W2h);
    // 3. raw attention scores (wave per edge)
    score_kernel<<<NEDGES / 8, 256, 0, stream>>>(q, k, ei, rel_bias, exbuf);
    // 4. per-chunk/head max
    chunkmax_kernel<<<NCHUNK * NHEAD, 256, 0, stream>>>(exbuf, chmax);
    // 5. exp + per-chunk/head sum (exbuf now holds exp values)
    chunksum_kernel<<<NCHUNK * NHEAD, 256, 0, stream>>>(exbuf, chmax, chsum);
    // 6. FiLM MLP via WMMA + softmax apply + scatter-add
    film_kernel<<<NEDGES / 64, 128, 0, stream>>>(v, ei, edge_attr, edge_time,
                                                 Wt, bt, b1, b2, W1h, W2h,
                                                 exbuf, chsum, out);
}
